// LongDistanceAttention_78091095376340
// MI455X (gfx1250) — compile-verified, matrix operand order checked by reading, not run on hardware
//
#include <hip/hip_runtime.h>
#include <math.h>

// ---------------------------------------------------------------------------
// Types for CDNA5 WMMA
// ---------------------------------------------------------------------------
typedef __attribute__((ext_vector_type(16))) __bf16 v16bf;
typedef __attribute__((ext_vector_type(8)))  float  v8f;

#define NN    6144
#define FIN   256
#define FOUT  128
#define NWORD 192      // 6144 bits / 32
#define MAXN  64       // cap on direct neighbors per row (E[nnz]=18.4, sd~4.3)

// float -> bf16 round-to-nearest-even
__device__ inline unsigned short f2bf(float f) {
  union { float f; unsigned u; } v; v.f = f;
  unsigned r = v.u + 0x7FFFu + ((v.u >> 16) & 1u);
  return (unsigned short)(r >> 16);
}

// Load a 16-bit A-style WMMA fragment (16x32) for this lane.
// p points at (row_base + k0) in u32 units (2 bf16 per u32).
// Lane layout (CDNA5 ISA, 16-bit A 16x32): lane = 16*khalf + row,
// VGPR g holds K = (g<4?0:16) + 8*khalf + 2*(g&3) + {0,1}.
__device__ inline v16bf frag_ld(const unsigned* p, int khalf) {
  union { v16bf v; unsigned u[8]; } r;
#pragma unroll
  for (int g = 0; g < 8; ++g) {
    int kb = ((g < 4) ? 0 : 16) + 8 * khalf + 2 * (g & 3);
    r.u[g] = p[kb >> 1];
  }
  return r.v;
}

__device__ inline v8f wmma_bf16(v16bf a, v16bf b, v8f c) {
  return __builtin_amdgcn_wmma_f32_16x16x32_bf16(false, a, false, b, (short)0, c,
                                                 false, false);
}

// ---------------------------------------------------------------------------
// Prep kernels
// ---------------------------------------------------------------------------
__global__ void to_bf16(const float* __restrict__ x, unsigned short* __restrict__ y, int n) {
  int i = blockIdx.x * blockDim.x + threadIdx.x;
  if (i < n) y[i] = f2bf(x[i]);
}

// W (K x N row-major f32) -> Wt (N x K row-major bf16)  == B^T for WMMA B-frags
__global__ void transpose_bf16(const float* __restrict__ W, unsigned short* __restrict__ Wt,
                               int K, int N) {
  int i = blockIdx.x * blockDim.x + threadIdx.x;
  if (i < K * N) {
    int k = i / N, n = i % N;
    Wt[(size_t)n * K + k] = f2bf(W[(size_t)k * N + n]);
  }
}

// ---------------------------------------------------------------------------
// Generic GEMM: C(M x 128) = A(M x K) * B(K x 128), A bf16 row-major,
// Bt = B^T bf16 row-major (128 x K). 8 waves/block, one 16x16 tile per wave.
// ---------------------------------------------------------------------------
__global__ void gemm_bf16_n128(const unsigned short* __restrict__ A,
                               const unsigned short* __restrict__ Bt,
                               float* __restrict__ C,
                               unsigned short* __restrict__ Cb,
                               int K) {
  int wave = threadIdx.x >> 5;
  int lane = threadIdx.x & 31;
  int r16 = lane & 15, half = lane >> 4;
  int rowbase = blockIdx.x * 16;
  int ntile = wave * 16;

  v8f acc = {0.f, 0.f, 0.f, 0.f, 0.f, 0.f, 0.f, 0.f};
  const unsigned* arow = (const unsigned*)(A + (size_t)(rowbase + r16) * K);
  const unsigned* brow = (const unsigned*)(Bt + (size_t)(ntile + r16) * K);
  for (int k0 = 0; k0 < K; k0 += 32) {
    v16bf a = frag_ld(arow + (k0 >> 1), half);
    v16bf b = frag_ld(brow + (k0 >> 1), half);
    acc = wmma_bf16(a, b, acc);
  }
  // D layout: lane holds (m = v + 8*half, n = r16)
#pragma unroll
  for (int v = 0; v < 8; ++v) {
    int m = v + 8 * half;
    size_t idx = (size_t)(rowbase + m) * FOUT + ntile + r16;
    if (C)  C[idx]  = acc[v];
    if (Cb) Cb[idx] = f2bf(acc[v]);
  }
}

// ---------------------------------------------------------------------------
// Row dots: s1 = Wh @ r[:128], s2 = Wh @ r[128:256].  One wave per row.
// ---------------------------------------------------------------------------
__global__ void row_dots(const float* __restrict__ Wh, const float* __restrict__ r,
                         float* __restrict__ s1, float* __restrict__ s2) {
  int row = blockIdx.x * (blockDim.x >> 5) + (threadIdx.x >> 5);
  int lane = threadIdx.x & 31;
  float p1 = 0.f, p2 = 0.f;
#pragma unroll
  for (int q = 0; q < 4; ++q) {
    int f = lane + 32 * q;
    float h = Wh[(size_t)row * FOUT + f];
    p1 += h * r[f];
    p2 += h * r[FOUT + f];
  }
  for (int m = 16; m; m >>= 1) {
    p1 += __shfl_xor(p1, m, 32);
    p2 += __shfl_xor(p2, m, 32);
  }
  if (lane == 0) { s1[row] = p1; s2[row] = p2; }
}

__global__ void col_sum(const float* __restrict__ Wh, float* __restrict__ cs) {
  int c = threadIdx.x;
  float s = 0.f;
  for (int i = 0; i < NN; ++i) s += Wh[(size_t)i * FOUT + c];
  cs[c] = s;
}

// ---------------------------------------------------------------------------
// Adjacency scan: read A once, build bitset rows + neighbor lists via ballot.
// One wave per row.
// ---------------------------------------------------------------------------
__global__ void adj_scan(const float* __restrict__ A, unsigned* __restrict__ bits,
                         int* __restrict__ nbr, int* __restrict__ cnt) {
  int row = blockIdx.x * (blockDim.x >> 5) + (threadIdx.x >> 5);
  int lane = threadIdx.x & 31;
  const float* arow = A + (size_t)row * NN;
  int c = 0;
  for (int w = 0; w < NWORD; ++w) {
    float a = arow[w * 32 + lane];
    unsigned m = __builtin_amdgcn_ballot_w32(a != 0.0f);
    if (lane == 0) {
      bits[(size_t)row * NWORD + w] = m;
      unsigned mm = m;
      while (mm) {
        int b = __builtin_ctz(mm);
        if (c < MAXN) nbr[(size_t)row * MAXN + c] = w * 32 + b;
        ++c;
        mm &= mm - 1;
      }
    }
  }
  if (lane == 0) cnt[row] = (c > MAXN) ? MAXN : c;
}

// dst[i] = OR over direct neighbors j of src[j]   (mask(A * src-mask))
__global__ void or_expand(const unsigned* __restrict__ src, unsigned* __restrict__ dst,
                          const int* __restrict__ nbr, const int* __restrict__ cnt) {
  int row = blockIdx.x;
  int t = threadIdx.x;            // 192 threads, one word each
  int n = cnt[row];
  unsigned acc = 0;
  for (int c = 0; c < n; ++c) {
    int j = nbr[(size_t)row * MAXN + c];
    acc |= src[(size_t)j * NWORD + t];
  }
  dst[(size_t)row * NWORD + t] = acc;
}

// ---------------------------------------------------------------------------
// Hop-1 (sparse): hk = gelu((colsum(Wh) + sum_{j in N(i)} (e^{e_ij}-1) Wh_j)/denom_i)
// with e_ij = leaky_relu(s1_i + s2_j, 0.2). One wave per row.
// Outputs hk in bf16 row-major and bf16 transposed (feature-major).
// ---------------------------------------------------------------------------
__global__ void hop1(const float* __restrict__ Wh, const float* __restrict__ s1,
                     const float* __restrict__ s2, const float* __restrict__ cs,
                     const int* __restrict__ nbr, const int* __restrict__ cnt,
                     unsigned short* __restrict__ hk_b, unsigned short* __restrict__ hk_t) {
  int row = blockIdx.x * (blockDim.x >> 5) + (threadIdx.x >> 5);
  int lane = threadIdx.x & 31;
  int n = cnt[row];
  float s1i = s1[row];

  float dp = 0.f;
  for (int t = lane; t < n; t += 32) {
    int j = nbr[(size_t)row * MAXN + t];
    float e = s1i + s2[j];
    e = e > 0.f ? e : 0.2f * e;
    dp += expf(e);
  }
  for (int m = 16; m; m >>= 1) dp += __shfl_xor(dp, m, 32);   // all lanes get sum

  float acc[4];
#pragma unroll
  for (int q = 0; q < 4; ++q) acc[q] = cs[lane + 32 * q];

  for (int t = 0; t < n; ++t) {
    int j = nbr[(size_t)row * MAXN + t];
    float e = s1i + s2[j];
    e = e > 0.f ? e : 0.2f * e;
    float w = expf(e) - 1.0f;
    const float* whj = Wh + (size_t)j * FOUT;
#pragma unroll
    for (int q = 0; q < 4; ++q) acc[q] += w * whj[lane + 32 * q];
  }

#pragma unroll
  for (int q = 0; q < 4; ++q) {
    float h = acc[q] / dp;
    float g = 0.5f * h * (1.0f + erff(h * 0.70710678118f));   // exact gelu
    int f = lane + 32 * q;
    unsigned short gb = f2bf(g);
    hk_b[(size_t)row * FOUT + f] = gb;
    hk_t[(size_t)f * NN + row]   = gb;
  }
}

// ---------------------------------------------------------------------------
// Flash hop-3: out_pre = softmax_{mask(A^3)}(hk @ Wa^T) @ hk
// One wave owns a 16-row tile; loop over 6144 columns in steps of 32.
// Per step: scores via WMMA, mask+exp -> LDS (bf16), reload as A-fragment,
// numerator WMMAs (w @ hk) + denominator WMMA (w @ ones).
// ---------------------------------------------------------------------------
#define FWAVES 4
__global__ void flash_hop3(const unsigned short* __restrict__ hk_b,   // 6144 x 128
                           const unsigned short* __restrict__ hk_t,   // 128 x 6144
                           const unsigned short* __restrict__ WaM_b,  // 6144 x 128
                           const unsigned* __restrict__ bits3,        // 6144 x 192
                           unsigned short* __restrict__ outpre) {     // 6144 x 128 bf16
  __shared__ unsigned short wstage[FWAVES][16 * 32];
  int wave = threadIdx.x >> 5;
  int lane = threadIdx.x & 31;
  int r16 = lane & 15, half = lane >> 4;
  int rowbase = (blockIdx.x * FWAVES + wave) * 16;

  // Hoisted hk A-fragments for this wave's 16 rows (K = 128 -> 4 frags)
  const unsigned* hkrow = (const unsigned*)(hk_b + (size_t)(rowbase + r16) * FOUT);
  v16bf ahk[4];
#pragma unroll
  for (int q = 0; q < 4; ++q) ahk[q] = frag_ld(hkrow + q * 16, half);

  v8f num[8];
#pragma unroll
  for (int t = 0; t < 8; ++t) num[t] = (v8f){0.f, 0.f, 0.f, 0.f, 0.f, 0.f, 0.f, 0.f};
  v8f den = {0.f, 0.f, 0.f, 0.f, 0.f, 0.f, 0.f, 0.f};

  union { v16bf v; unsigned u[8]; } onesu;
#pragma unroll
  for (int g = 0; g < 8; ++g) onesu.u[g] = 0x3F803F80u;   // bf16 1.0 pairs
  v16bf ones = onesu.v;

  unsigned short* wl = &wstage[wave][0];

  for (int j = 0; j < NN; j += 32) {
    // ---- scores for cols [j, j+32): two 16x16 tiles
    v8f sc0 = {0.f, 0.f, 0.f, 0.f, 0.f, 0.f, 0.f, 0.f};
    v8f sc1 = {0.f, 0.f, 0.f, 0.f, 0.f, 0.f, 0.f, 0.f};
    const unsigned* b0r = (const unsigned*)(WaM_b + (size_t)(j + r16) * FOUT);
    const unsigned* b1r = (const unsigned*)(WaM_b + (size_t)(j + 16 + r16) * FOUT);
    if (j + 32 < NN)
      __builtin_prefetch((const void*)(WaM_b + (size_t)(j + 32 + r16) * FOUT), 0, 1);
#pragma unroll
    for (int q = 0; q < 4; ++q) {
      v16bf b0 = frag_ld(b0r + q * 16, half);
      v16bf b1 = frag_ld(b1r + q * 16, half);
      sc0 = wmma_bf16(ahk[q], b0, sc0);
      sc1 = wmma_bf16(ahk[q], b1, sc1);
    }

    // ---- mask by bits3, exponentiate, stage to LDS as bf16 w[16][32]
    int wordidx = j >> 5;
#pragma unroll
    for (int v = 0; v < 8; ++v) {
      int m = v + 8 * half;
      unsigned bw = bits3[(size_t)(rowbase + m) * NWORD + wordidx];
      float w0 = ((bw >> r16) & 1u)        ? expf(sc0[v]) : 0.f;
      float w1 = ((bw >> (16 + r16)) & 1u) ? expf(sc1[v]) : 0.f;
      wl[m * 32 + r16]      = f2bf(w0);
      wl[m * 32 + 16 + r16] = f2bf(w1);
    }
    asm volatile("s_wait_dscnt 0" ::: "memory");

    // ---- reload weights as A-fragment (16x32)
    v16bf aw = frag_ld((const unsigned*)(wl + r16 * 32), half);

    // ---- numerator: num(16x128) += w(16x32) @ hk(32x128); B^T rows = hk_t rows
#pragma unroll
    for (int t = 0; t < 8; ++t) {
      const unsigned* hr = (const unsigned*)(hk_t + (size_t)(16 * t + r16) * NN) + (j >> 1);
      v16bf bh = frag_ld(hr, half);
      num[t] = wmma_bf16(aw, bh, num[t]);
    }
    // ---- denominator: den += w @ ones (every column = row sum)
    den = wmma_bf16(aw, ones, den);
  }

  // ---- epilogue: out_pre[m][f] = num/den  (bf16 for the final WMMA GEMM)
#pragma unroll
  for (int t = 0; t < 8; ++t) {
#pragma unroll
    for (int v = 0; v < 8; ++v) {
      int m = v + 8 * half;
      int f = 16 * t + r16;
      float o = num[t][v] / den[v];
      outpre[(size_t)(rowbase + m) * FOUT + f] = f2bf(o);
    }
  }
}

// ---------------------------------------------------------------------------
// Host-side launch
// ---------------------------------------------------------------------------
extern "C" void kernel_launch(void* const* d_in, const int* in_sizes, int n_in,
                              void* d_out, int out_size, void* d_ws, size_t ws_size,
                              hipStream_t stream) {
  (void)in_sizes; (void)n_in; (void)out_size; (void)ws_size;
  const float* X       = (const float*)d_in[0];
  const float* A       = (const float*)d_in[1];
  const float* W_short = (const float*)d_in[2];
  const float* r       = (const float*)d_in[3];
  const float* W_a     = (const float*)d_in[4];
  const float* W_out   = (const float*)d_in[5];
  float* out = (float*)d_out;

  char* ws = (char*)d_ws;
  size_t off = 0;
  auto alloc = [&](size_t bytes) -> char* {
    char* p = ws + off;
    off = (off + bytes + 255) & ~(size_t)255;
    return p;
  };

  unsigned short* Xb     = (unsigned short*)alloc((size_t)NN * FIN * 2);
  unsigned short* Wst    = (unsigned short*)alloc((size_t)FOUT * FIN * 2);
  unsigned short* Wat    = (unsigned short*)alloc((size_t)FOUT * FIN * 2);
  unsigned short* Woutt  = (unsigned short*)alloc((size_t)FOUT * FOUT * 2);
  float*          Wh     = (float*)alloc((size_t)NN * FOUT * 4);
  unsigned short* WaMb   = (unsigned short*)alloc((size_t)NN * FOUT * 2);
  float*          s1     = (float*)alloc((size_t)NN * 4);
  float*          s2     = (float*)alloc((size_t)NN * 4);
  float*          cs     = (float*)alloc((size_t)FOUT * 4);
  unsigned*       bits1  = (unsigned*)alloc((size_t)NN * NWORD * 4);
  unsigned*       bits2  = (unsigned*)alloc((size_t)NN * NWORD * 4);
  unsigned*       bits3  = (unsigned*)alloc((size_t)NN * NWORD * 4);
  int*            nbr    = (int*)alloc((size_t)NN * MAXN * 4);
  int*            cnt    = (int*)alloc((size_t)NN * 4);
  unsigned short* hkb    = (unsigned short*)alloc((size_t)NN * FOUT * 2);
  unsigned short* hkt    = (unsigned short*)alloc((size_t)FOUT * NN * 2);
  unsigned short* outpre = (unsigned short*)alloc((size_t)NN * FOUT * 2);

  // prep: bf16 conversions / transposes
  to_bf16<<<(NN * FIN + 255) / 256, 256, 0, stream>>>(X, Xb, NN * FIN);
  transpose_bf16<<<(FIN * FOUT + 255) / 256, 256, 0, stream>>>(W_short, Wst, FIN, FOUT);
  transpose_bf16<<<(FIN * FOUT + 255) / 256, 256, 0, stream>>>(W_a, Wat, FIN, FOUT);
  transpose_bf16<<<(FOUT * FOUT + 255) / 256, 256, 0, stream>>>(W_out, Woutt, FOUT, FOUT);

  // Wh = X @ W_short (f32 out), WaM = X @ W_a (bf16 out)
  gemm_bf16_n128<<<NN / 16, 256, 0, stream>>>(Xb, Wst, Wh, nullptr, FIN);
  gemm_bf16_n128<<<NN / 16, 256, 0, stream>>>(Xb, Wat, nullptr, WaMb, FIN);

  // scalars
  row_dots<<<NN / 4, 128, 0, stream>>>(Wh, r, s1, s2);
  col_sum<<<1, 128, 0, stream>>>(Wh, cs);

  // masks: A read exactly once; A^2/A^3 masks by bitset OR over neighbor lists
  adj_scan<<<NN / 4, 128, 0, stream>>>(A, bits1, nbr, cnt);
  or_expand<<<NN, NWORD, 0, stream>>>(bits1, bits2, nbr, cnt);
  or_expand<<<NN, NWORD, 0, stream>>>(bits2, bits3, nbr, cnt);

  // hop-1 sparse attention -> hk (bf16 row-major + transposed)
  hop1<<<NN / 4, 128, 0, stream>>>(Wh, s1, s2, cs, nbr, cnt, hkb, hkt);

  // fused scores + masked softmax + AV (flash style, all WMMA)
  flash_hop3<<<NN / 16 / FWAVES, FWAVES * 32, 0, stream>>>(hkb, hkt, WaMb, bits3, outpre);

  // final projection: out = out_pre @ W_out
  gemm_bf16_n128<<<NN / 16, 256, 0, stream>>>(outpre, Woutt, out, nullptr, FOUT);
}
